// VideoSwinBasicLayer_59631325938166
// MI455X (gfx1250) — compile-verified, matrix operand order checked by reading, not run on hardware
//
#include <hip/hip_runtime.h>
#include <stdint.h>

// ---------------- types ----------------
typedef __bf16 bf16;
typedef __attribute__((ext_vector_type(16))) __bf16 v16bf;
typedef __attribute__((ext_vector_type(8)))  __bf16 v8bf;
typedef __attribute__((ext_vector_type(8)))  float  v8f;
typedef __attribute__((ext_vector_type(4)))  float  v4f;
typedef unsigned int u32x4 __attribute__((ext_vector_type(4)));
typedef int i32x8 __attribute__((ext_vector_type(8)));
typedef int i32x4 __attribute__((ext_vector_type(4)));

// ---------------- static config ----------------
#define DDIM 16
#define HDIMG 56
#define WDIMG 56
#define CC 128
#define NHEAD 4
#define HDHEAD 32
#define WSD 8
#define WSH 7
#define WSW 7
#define LWIN 392          // tokens per window
#define LP 416            // padded to 13*32
#define NWIN 128          // (16/8)*(56/7)*(56/7)
#define TOK (DDIM*HDIMG*WDIMG)   // 50176
#define HID 512
#define TBL 2535
#define ATT_SCALE 0.17677669529663687f   // 32^-0.5

// ---------------- helpers ----------------
__device__ __forceinline__ v8f zero8() {
  v8f z;
#pragma unroll
  for (int i = 0; i < 8; ++i) z[i] = 0.0f;
  return z;
}

__device__ __forceinline__ v8f wmma_bf16(v16bf a, v16bf b, v8f c) {
  return __builtin_amdgcn_wmma_f32_16x16x32_bf16(false, a, false, b, (short)0, c,
                                                 false, false);
}

// Load a 16-row x 32-K tile (row-major, stride ld) into the CDNA5 WMMA
// 16-bit A/B register layout:
//   lanes 0-15  : row = lane,    K chunks {0..7} and {16..23}
//   lanes 16-31 : row = lane-16, K chunks {8..15} and {24..31}
__device__ __forceinline__ v16bf load_tile(const bf16* __restrict__ base, int ld) {
  const int lane = threadIdx.x & 31;
  const int row  = lane & 15;
  const int kc   = (lane < 16) ? 0 : 8;
  const bf16* p = base + (size_t)row * ld + kc;
  v8bf c0 = *(const v8bf*)(p);
  v8bf c1 = *(const v8bf*)(p + 16);
  v16bf r;
#pragma unroll
  for (int i = 0; i < 8; ++i) { r[i] = c0[i]; r[i + 8] = c1[i]; }
  return r;
}

// Same tile gather but from an LDS-resident panel.
__device__ __forceinline__ v16bf lds_tile(const bf16* base, int ld) {
  const int lane = threadIdx.x & 31;
  const int row  = lane & 15;
  const int kc   = (lane < 16) ? 0 : 8;
  const bf16* p = base + (size_t)row * ld + kc;
  v8bf c0 = *(const v8bf*)(p);
  v8bf c1 = *(const v8bf*)(p + 16);
  v16bf r;
#pragma unroll
  for (int i = 0; i < 8; ++i) { r[i] = c0[i]; r[i + 8] = c1[i]; }
  return r;
}

// One 16-byte async global -> LDS copy issued by this lane (ASYNCcnt tracked).
__device__ __forceinline__ void async_b128_to_lds(bf16* lds_dst, const bf16* gsrc) {
  uint32_t lo = (uint32_t)(uintptr_t)lds_dst;       // flat LDS aperture: [31:0] = offset
  uint64_t ga = (uint64_t)(uintptr_t)gsrc;
  asm volatile("global_load_async_to_lds_b128 %0, %1, off"
               :: "v"(lo), "v"(ga) : "memory");
}
__device__ __forceinline__ void wait_async0() {
  asm volatile("s_wait_asynccnt 0" ::: "memory");
}

// TDM: DMA a contiguous byte range (multiple of 8B) global -> LDS as a
// 1-row tensor tile. User-mode D#: count=1, type=2("image"), data_size=8B,
// tensor_dim0 = tile_dim0 = bytes/8, tensor_dim1 = tile_dim1 = 1.
__device__ __forceinline__ void tdm_load_to_lds(bf16* lds_dst, const bf16* gsrc,
                                                unsigned bytes) {
  unsigned units = bytes >> 3;                       // 8-byte elements
  unsigned lds_addr = (uint32_t)(uintptr_t)lds_dst;
  unsigned long long ga = (unsigned long long)(uintptr_t)gsrc;
  u32x4 g0;
  g0[0] = 1u;                                        // count=1 (valid descriptor)
  g0[1] = lds_addr;                                  // lds_addr [63:32]
  g0[2] = (unsigned)(ga & 0xffffffffu);              // global_addr low
  g0[3] = (unsigned)((ga >> 32) & 0x01ffffffu)       // global_addr [56:32]
          | (2u << 30);                              // type=2
  i32x8 g1;
  g1[0] = (int)(3u << 16);                           // data_size=3 (8B), no masks
  g1[1] = (int)((units & 0xffffu) << 16);            // tensor_dim0 [15:0]
  g1[2] = (int)((units >> 16) | (1u << 16));         // tensor_dim0 hi | tensor_dim1=1
  g1[3] = (int)((units & 0xffffu) << 16);            // tile_dim0
  g1[4] = 1;                                         // tile_dim1=1, tile_dim2=0
  g1[5] = (int)units;                                // tensor_dim0_stride low
  g1[6] = 0;
  g1[7] = 0;
  i32x4 z4 = {0, 0, 0, 0};
#if __clang_major__ >= 23
  i32x8 z8 = {0, 0, 0, 0, 0, 0, 0, 0};
  __builtin_amdgcn_tensor_load_to_lds(g0, g1, z4, z4, z8, 0);
#else
  __builtin_amdgcn_tensor_load_to_lds(g0, g1, z4, z4, 0);
#endif
}

__device__ __forceinline__ float wave_sum32(float v) {
#pragma unroll
  for (int o = 16; o >= 1; o >>= 1) v += __shfl_xor(v, o, 32);
  return v;
}
__device__ __forceinline__ float half_max16(float v) {
#pragma unroll
  for (int o = 8; o >= 1; o >>= 1) v = fmaxf(v, __shfl_xor(v, o, 32));
  return v;
}
__device__ __forceinline__ float half_sum16(float v) {
#pragma unroll
  for (int o = 8; o >= 1; o >>= 1) v += __shfl_xor(v, o, 32);
  return v;
}
__device__ __forceinline__ int region3(int v, int a, int b) {
  return (v < a) ? 0 : ((v < b) ? 1 : 2);
}

// Shared GEMM inner machinery: stage a 64-column x KD weight panel (contiguous
// (N,K) bf16) into LDS asynchronously, then accumulate a 16x64 wave tile.
// B tiles are loaded into distinct values ahead of the WMMAs so the scheduler
// can issue the ds_load_b128s back-to-back and use partial DScnt waits.
template <int KD>
__device__ __forceinline__ void gemm_panel(const bf16* __restrict__ A,
                                           const bf16* __restrict__ Bpanel,
                                           bf16* smem, v8f acc[4]) {
  const int tid = threadIdx.x;
  constexpr int CHUNKS = (64 * KD) / 8;   // 16B chunks
#pragma unroll 1
  for (int c = tid; c < CHUNKS; c += 256)
    async_b128_to_lds(smem + c * 8, Bpanel + c * 8);
  __builtin_prefetch(A, 0, 0);            // global_prefetch for this wave's A rows
  wait_async0();
  __syncthreads();
#pragma unroll
  for (int ks = 0; ks < KD / 32; ++ks) {
    v16bf a  = load_tile(A + ks * 32, KD);
    v16bf b0 = lds_tile(smem + (size_t)(0 * 16) * KD + ks * 32, KD);
    v16bf b1 = lds_tile(smem + (size_t)(1 * 16) * KD + ks * 32, KD);
    v16bf b2 = lds_tile(smem + (size_t)(2 * 16) * KD + ks * 32, KD);
    v16bf b3 = lds_tile(smem + (size_t)(3 * 16) * KD + ks * 32, KD);
    acc[0] = wmma_bf16(a, b0, acc[0]);
    acc[1] = wmma_bf16(a, b1, acc[1]);
    acc[2] = wmma_bf16(a, b2, acc[2]);
    acc[3] = wmma_bf16(a, b3, acc[3]);
  }
}

// ---------------- weight transpose + bf16 convert ----------------
// src (K,N) fp32 row-major -> dst (N,K) bf16 row-major
__global__ void k_transpose_bf16(const float* __restrict__ src, bf16* __restrict__ dst,
                                 int K, int N) {
  int idx = blockIdx.x * 256 + threadIdx.x;
  if (idx >= K * N) return;
  int n = idx / K;
  int k = idx - n * K;
  dst[idx] = (bf16)src[(size_t)k * N + n];
}

// ---------------- LN1 + roll + window partition (fp32 -> bf16) --------------
__global__ void k_ln_partition(const float* __restrict__ x, const float* __restrict__ g,
                               const float* __restrict__ bb, bf16* __restrict__ xw,
                               int sd, int sh, int sw) {
  int wave = threadIdx.x >> 5, lane = threadIdx.x & 31;
  int token = blockIdx.x * 4 + wave;
  int win = token / LP, t = token - win * LP;
  bf16* dst = xw + (size_t)token * CC + lane * 4;
  if (t >= LWIN) {
#pragma unroll
    for (int i = 0; i < 4; ++i) dst[i] = (bf16)0.0f;
    return;
  }
  int wd = win >> 6, wh = (win >> 3) & 7, ww = win & 7;
  int td = t / 49, rem = t - td * 49, th = rem / 7, tw = rem - th * 7;
  int d = wd * WSD + td + sd; if (d >= DDIM)  d -= DDIM;
  int h = wh * WSH + th + sh; if (h >= HDIMG) h -= HDIMG;
  int w = ww * WSW + tw + sw; if (w >= WDIMG) w -= WDIMG;
  const float* row = x + (size_t)((d * HDIMG + h) * WDIMG + w) * CC;
  v4f v = *(const v4f*)(row + lane * 4);
  float s = v[0] + v[1] + v[2] + v[3];
  float mu = wave_sum32(s) * (1.0f / CC);
  float qs = 0.0f;
#pragma unroll
  for (int i = 0; i < 4; ++i) { float dl = v[i] - mu; qs += dl * dl; }
  float inv = rsqrtf(wave_sum32(qs) * (1.0f / CC) + 1e-5f);
#pragma unroll
  for (int i = 0; i < 4; ++i) {
    int ch = lane * 4 + i;
    dst[i] = (bf16)((v[i] - mu) * inv * g[ch] + bb[ch]);
  }
}

// ---------------- QKV GEMM: (NWIN*LP,128) x (128,384), scatter q/k/vT -------
__global__ void k_qkv(const bf16* __restrict__ xw, const bf16* __restrict__ wT,
                      const float* __restrict__ bias, bf16* __restrict__ q,
                      bf16* __restrict__ kk, bf16* __restrict__ vt) {
  __shared__ __align__(16) bf16 bsm[64 * CC];
  int wave = threadIdx.x >> 5, lane = threadIdx.x & 31;
  int Mb = blockIdx.x * 128 + wave * 16;
  int Nb = blockIdx.y * 64;
  v8f acc[4] = {zero8(), zero8(), zero8(), zero8()};
  gemm_panel<CC>(xw + (size_t)Mb * CC, wT + (size_t)Nb * CC, bsm, acc);
  int mbase = (lane < 16) ? 0 : 8;
  int nloc = lane & 15;
#pragma unroll
  for (int nt = 0; nt < 4; ++nt) {
#pragma unroll
    for (int r = 0; r < 8; ++r) {
      int m = Mb + mbase + r;
      int gcol = Nb + nt * 16 + nloc;
      float val = acc[nt][r] + bias[gcol];
      int win = m / LP, t = m - win * LP;
      int which = gcol >> 7;       // 0=q 1=k 2=v
      int c = gcol & 127;
      int head = c >> 5, hc = c & 31;
      size_t hb = (size_t)(win * NHEAD + head);
      if (which == 0)       q[(hb * LP + t) * HDHEAD + hc] = (bf16)(val * ATT_SCALE);
      else if (which == 1) kk[(hb * LP + t) * HDHEAD + hc] = (bf16)val;
      else                 vt[(hb * HDHEAD + hc) * LP + t] = (bf16)val;  // transposed
    }
  }
}

// ---------------- fused flash attention per (window, head, 16-row tile) -----
__global__ void k_attn(const bf16* __restrict__ q, const bf16* __restrict__ kk,
                       const bf16* __restrict__ vt, const float* __restrict__ rpb,
                       bf16* __restrict__ out, int shifted) {
  __shared__ __align__(16) bf16 ksm[LP * HDHEAD];     // 26,624 B
  __shared__ __align__(16) bf16 vsm[HDHEAD * LP];     // 26,624 B
  __shared__ __align__(16) bf16 pbuf[8][16 * 32];     //  8,192 B
  int wave = threadIdx.x >> 5, lane = threadIdx.x & 31;
  int win = blockIdx.x, head = blockIdx.y;
  int Mb = blockIdx.z * 128 + wave * 16;
  size_t hb = (size_t)(win * NHEAD + head);
  const bf16* Kg = kk + hb * LP * HDHEAD;
  const bf16* Vg = vt + hb * HDHEAD * LP;

  // Tensor Data Mover stages the whole K and V^T panels for this (win, head):
  // one DMA per panel, issued by wave 0, tracked by TENSORcnt.
  if (wave == 0) {
    tdm_load_to_lds(ksm, Kg, LP * HDHEAD * sizeof(bf16));
    tdm_load_to_lds(vsm, Vg, HDHEAD * LP * sizeof(bf16));
    __builtin_amdgcn_s_wait_tensorcnt(0);
  }
  __syncthreads();

  if (Mb < LP) {
    const bf16* Qp = q + (hb * LP + Mb) * HDHEAD;
    v16bf Qa = load_tile(Qp, HDHEAD);

    int wd = win >> 6, wh = (win >> 3) & 7, ww = win & 7;
    int nloc = lane & 15;
    int mbase = (lane < 16) ? 0 : 8;
    int itd[8], ith[8], itw[8], ilab[8];
#pragma unroll
    for (int r = 0; r < 8; ++r) {
      int i = Mb + mbase + r; if (i >= LWIN) i = LWIN - 1;
      int td = i / 49, rem = i - td * 49, th = rem / 7, tw = rem - th * 7;
      itd[r] = td; ith[r] = th; itw[r] = tw;
      ilab[r] = region3(wd * WSD + td, 8, 12) * 9 +
                region3(wh * WSH + th, 49, 53) * 3 +
                region3(ww * WSW + tw, 49, 53);
    }

    v8f O0 = zero8(), O1 = zero8();
    float mrow[8], lrow[8];
#pragma unroll
    for (int r = 0; r < 8; ++r) { mrow[r] = -1e30f; lrow[r] = 0.0f; }

    bf16* pb = pbuf[wave];
    for (int kb0 = 0; kb0 < LP; kb0 += 32) {
      v16bf Kb0 = lds_tile(ksm + (size_t)kb0 * HDHEAD, HDHEAD);
      v16bf Kb1 = lds_tile(ksm + (size_t)(kb0 + 16) * HDHEAD, HDHEAD);
      v16bf Vb0 = lds_tile(vsm + kb0, LP);                     // out ch 0-15
      v16bf Vb1 = lds_tile(vsm + (size_t)16 * LP + kb0, LP);   // out ch 16-31
      v8f S0 = wmma_bf16(Qa, Kb0, zero8());
      v8f S1 = wmma_bf16(Qa, Kb1, zero8());

      int j0 = kb0 + nloc, j1 = kb0 + 16 + nloc;
      float jpen0 = (j0 >= LWIN) ? -1e30f : 0.0f;
      float jpen1 = (j1 >= LWIN) ? -1e30f : 0.0f;
      int j0c = (j0 < LWIN) ? j0 : LWIN - 1;
      int j1c = (j1 < LWIN) ? j1 : LWIN - 1;
      int jtd0 = j0c / 49, jr0 = j0c - jtd0 * 49, jth0 = jr0 / 7, jtw0 = jr0 - jth0 * 7;
      int jtd1 = j1c / 49, jr1 = j1c - jtd1 * 49, jth1 = jr1 / 7, jtw1 = jr1 - jth1 * 7;
      int jlab0 = region3(wd * WSD + jtd0, 8, 12) * 9 +
                  region3(wh * WSH + jth0, 49, 53) * 3 +
                  region3(ww * WSW + jtw0, 49, 53);
      int jlab1 = region3(wd * WSD + jtd1, 8, 12) * 9 +
                  region3(wh * WSH + jth1, 49, 53) * 3 +
                  region3(ww * WSW + jtw1, 49, 53);

#pragma unroll
      for (int r = 0; r < 8; ++r) {
        int rel0 = (itd[r] - jtd0 + 7) * 169 + (ith[r] - jth0 + 6) * 13 + (itw[r] - jtw0 + 6);
        int rel1 = (itd[r] - jtd1 + 7) * 169 + (ith[r] - jth1 + 6) * 13 + (itw[r] - jtw1 + 6);
        float b0 = rpb[rel0 * NHEAD + head];
        float b1 = rpb[rel1 * NHEAD + head];
        float mk0 = (shifted && (ilab[r] != jlab0)) ? -100.0f : 0.0f;
        float mk1 = (shifted && (ilab[r] != jlab1)) ? -100.0f : 0.0f;
        float s0 = S0[r] + b0 + mk0 + jpen0;
        float s1 = S1[r] + b1 + mk1 + jpen1;
        float mx = half_max16(fmaxf(s0, s1));
        float nm = fmaxf(mrow[r], mx);
        float sc = __expf(mrow[r] - nm);
        float p0 = __expf(s0 - nm);
        float p1 = __expf(s1 - nm);
        float rs = half_sum16(p0 + p1);
        lrow[r] = lrow[r] * sc + rs;
        mrow[r] = nm;
        O0[r] *= sc;
        O1[r] *= sc;
        int mm = mbase + r;
        pb[mm * 32 + nloc]      = (bf16)p0;
        pb[mm * 32 + 16 + nloc] = (bf16)p1;
      }
      asm volatile("s_wait_dscnt 0" ::: "memory");   // same-wave LDS store->load
      int prow = lane & 15;
      int pkc = (lane < 16) ? 0 : 8;
      v8bf pc0 = *(const v8bf*)&pb[prow * 32 + pkc];
      v8bf pc1 = *(const v8bf*)&pb[prow * 32 + pkc + 16];
      v16bf Pa;
#pragma unroll
      for (int i2 = 0; i2 < 8; ++i2) { Pa[i2] = pc0[i2]; Pa[i2 + 8] = pc1[i2]; }
      O0 = wmma_bf16(Pa, Vb0, O0);
      O1 = wmma_bf16(Pa, Vb1, O1);
      asm volatile("s_wait_dscnt 0" ::: "memory");   // drain before next P store
    }

#pragma unroll
    for (int r = 0; r < 8; ++r) {
      float inv = 1.0f / (lrow[r] + 1e-20f);
      int m = Mb + mbase + r;
      size_t off = ((size_t)win * LP + m) * CC + head * HDHEAD;
      out[off + nloc]      = (bf16)(O0[r] * inv);
      out[off + 16 + nloc] = (bf16)(O1[r] * inv);
    }
  }
}

// ---------------- proj GEMM + window-reverse + unroll + residual add --------
__global__ void k_proj(const bf16* __restrict__ attn, const bf16* __restrict__ wT,
                       const float* __restrict__ bias, float* __restrict__ x,
                       int sd, int sh, int sw) {
  __shared__ __align__(16) bf16 bsm[64 * CC];
  int wave = threadIdx.x >> 5, lane = threadIdx.x & 31;
  int Mb = blockIdx.x * 128 + wave * 16;
  int Nb = blockIdx.y * 64;
  v8f acc[4] = {zero8(), zero8(), zero8(), zero8()};
  gemm_panel<CC>(attn + (size_t)Mb * CC, wT + (size_t)Nb * CC, bsm, acc);
  int mbase = (lane < 16) ? 0 : 8, nloc = lane & 15;
#pragma unroll
  for (int nt = 0; nt < 4; ++nt) {
#pragma unroll
    for (int r = 0; r < 8; ++r) {
      int m = Mb + mbase + r;
      int win = m / LP, t = m - win * LP;
      if (t >= LWIN) continue;
      int wd = win >> 6, wh = (win >> 3) & 7, ww = win & 7;
      int td = t / 49, rem = t - td * 49, th = rem / 7, tw = rem - th * 7;
      int d = wd * WSD + td + sd; if (d >= DDIM)  d -= DDIM;
      int h = wh * WSH + th + sh; if (h >= HDIMG) h -= HDIMG;
      int w = ww * WSW + tw + sw; if (w >= WDIMG) w -= WDIMG;
      int col = Nb + nt * 16 + nloc;
      size_t off = ((size_t)(d * HDIMG + h) * WDIMG + w) * CC + col;
      x[off] += acc[nt][r] + bias[col];   // unique (d,h,w,col) per element
    }
  }
}

// ---------------- LN2 (fp32 -> bf16, flat token order) ----------------------
__global__ void k_ln2(const float* __restrict__ x, const float* __restrict__ g,
                      const float* __restrict__ bb, bf16* __restrict__ h1) {
  int wave = threadIdx.x >> 5, lane = threadIdx.x & 31;
  int token = blockIdx.x * 4 + wave;
  const float* row = x + (size_t)token * CC;
  v4f v = *(const v4f*)(row + lane * 4);
  float s = v[0] + v[1] + v[2] + v[3];
  float mu = wave_sum32(s) * (1.0f / CC);
  float qs = 0.0f;
#pragma unroll
  for (int i = 0; i < 4; ++i) { float dl = v[i] - mu; qs += dl * dl; }
  float inv = rsqrtf(wave_sum32(qs) * (1.0f / CC) + 1e-5f);
  bf16* dst = h1 + (size_t)token * CC + lane * 4;
#pragma unroll
  for (int i = 0; i < 4; ++i) {
    int ch = lane * 4 + i;
    dst[i] = (bf16)((v[i] - mu) * inv * g[ch] + bb[ch]);
  }
}

// ---------------- FC1 GEMM + exact GELU (bf16 out) --------------------------
__global__ void k_fc1(const bf16* __restrict__ h1, const bf16* __restrict__ wT,
                      const float* __restrict__ bias, bf16* __restrict__ o) {
  __shared__ __align__(16) bf16 bsm[64 * CC];
  int wave = threadIdx.x >> 5, lane = threadIdx.x & 31;
  int Mb = blockIdx.x * 128 + wave * 16;
  int Nb = blockIdx.y * 64;
  v8f acc[4] = {zero8(), zero8(), zero8(), zero8()};
  gemm_panel<CC>(h1 + (size_t)Mb * CC, wT + (size_t)Nb * CC, bsm, acc);
  int mbase = (lane < 16) ? 0 : 8, nloc = lane & 15;
#pragma unroll
  for (int nt = 0; nt < 4; ++nt) {
#pragma unroll
    for (int r = 0; r < 8; ++r) {
      int m = Mb + mbase + r;
      int col = Nb + nt * 16 + nloc;
      float v = acc[nt][r] + bias[col];
      float gl = 0.5f * v * (1.0f + erff(v * 0.7071067811865476f));
      o[(size_t)m * HID + col] = (bf16)gl;
    }
  }
}

// ---------------- FC2 GEMM + residual add (fp32 x in place) -----------------
__global__ void k_fc2(const bf16* __restrict__ o, const bf16* __restrict__ wT,
                      const float* __restrict__ bias, float* __restrict__ x) {
  __shared__ __align__(16) bf16 bsm[64 * HID];
  int wave = threadIdx.x >> 5, lane = threadIdx.x & 31;
  int Mb = blockIdx.x * 128 + wave * 16;
  int Nb = blockIdx.y * 64;
  v8f acc[4] = {zero8(), zero8(), zero8(), zero8()};
  gemm_panel<HID>(o + (size_t)Mb * HID, wT + (size_t)Nb * HID, bsm, acc);
  int mbase = (lane < 16) ? 0 : 8, nloc = lane & 15;
#pragma unroll
  for (int nt = 0; nt < 4; ++nt) {
#pragma unroll
    for (int r = 0; r < 8; ++r) {
      int m = Mb + mbase + r;
      int col = Nb + nt * 16 + nloc;
      x[(size_t)m * CC + col] += acc[nt][r] + bias[col];
    }
  }
}

// ---------------- host-side orchestration ----------------------------------
extern "C" void kernel_launch(void* const* d_in, const int* in_sizes, int n_in,
                              void* d_out, int out_size, void* d_ws, size_t ws_size,
                              hipStream_t stream) {
  (void)in_sizes; (void)n_in; (void)out_size; (void)ws_size;
  const float* x_in   = (const float*)d_in[0];
  const float* ln1_g  = (const float*)d_in[1];
  const float* ln1_b  = (const float*)d_in[2];
  const float* qkv_w  = (const float*)d_in[3];
  const float* qkv_b  = (const float*)d_in[4];
  const float* rpb    = (const float*)d_in[5];
  const float* proj_w = (const float*)d_in[6];
  const float* proj_b = (const float*)d_in[7];
  const float* ln2_g  = (const float*)d_in[8];
  const float* ln2_b  = (const float*)d_in[9];
  const float* fc1_w  = (const float*)d_in[10];
  const float* fc1_b  = (const float*)d_in[11];
  const float* fc2_w  = (const float*)d_in[12];
  const float* fc2_b  = (const float*)d_in[13];
  float* x = (float*)d_out;
  char* ws = (char*)d_ws;

  const size_t S1 = (size_t)NWIN * LP * CC * sizeof(bf16);   // 13,631,488 B
  bf16* wt_qkv  = (bf16*)(ws);
  bf16* wt_proj = (bf16*)(ws + 196608);
  bf16* wt_fc1  = (bf16*)(ws + 262144);
  bf16* wt_fc2  = (bf16*)(ws + 524288);
  char* base = ws + (1 << 20);
  bf16* xw  = (bf16*)(base);            // LN1 output, window space
  bf16* qb  = (bf16*)(base + S1);
  bf16* kb  = (bf16*)(base + 2 * S1);
  bf16* vtb = (bf16*)(base + 3 * S1);
  bf16* ab  = (bf16*)(base + 4 * S1);   // attention output (end ~69 MB)
  bf16* h1  = xw;                       // reuse: LN2 output
  bf16* f1  = qb;                       // reuse: FC1 output (50176*512*2 B)

  hipMemcpyAsync(x, x_in, (size_t)TOK * CC * sizeof(float),
                 hipMemcpyDeviceToDevice, stream);

  for (int blk = 0; blk < 2; ++blk) {
    k_transpose_bf16<<<(384 * 128 + 255) / 256, 256, 0, stream>>>(
        qkv_w + (size_t)blk * CC * 3 * CC, wt_qkv + (size_t)blk * 384 * CC, CC, 384);
    k_transpose_bf16<<<(128 * 128 + 255) / 256, 256, 0, stream>>>(
        proj_w + (size_t)blk * CC * CC, wt_proj + (size_t)blk * CC * CC, CC, CC);
    k_transpose_bf16<<<(512 * 128 + 255) / 256, 256, 0, stream>>>(
        fc1_w + (size_t)blk * CC * HID, wt_fc1 + (size_t)blk * HID * CC, CC, HID);
    k_transpose_bf16<<<(512 * 128 + 255) / 256, 256, 0, stream>>>(
        fc2_w + (size_t)blk * HID * CC, wt_fc2 + (size_t)blk * CC * HID, HID, CC);
  }

  for (int blk = 0; blk < 2; ++blk) {
    const int sd = blk ? 4 : 0, sh = blk ? 3 : 0, sw = blk ? 3 : 0;
    k_ln_partition<<<NWIN * LP / 4, 128, 0, stream>>>(
        x, ln1_g + blk * CC, ln1_b + blk * CC, xw, sd, sh, sw);
    k_qkv<<<dim3(NWIN * LP / 128, 6), 256, 0, stream>>>(
        xw, wt_qkv + (size_t)blk * 384 * CC, qkv_b + blk * 384, qb, kb, vtb);
    k_attn<<<dim3(NWIN, NHEAD, 4), 256, 0, stream>>>(
        qb, kb, vtb, rpb + (size_t)blk * TBL * NHEAD, ab, blk);
    k_proj<<<dim3(NWIN * LP / 128, 2), 256, 0, stream>>>(
        ab, wt_proj + (size_t)blk * CC * CC, proj_b + blk * CC, x, sd, sh, sw);
    k_ln2<<<TOK / 4, 128, 0, stream>>>(x, ln2_g + blk * CC, ln2_b + blk * CC, h1);
    k_fc1<<<dim3(TOK / 128, 8), 256, 0, stream>>>(
        h1, wt_fc1 + (size_t)blk * HID * CC, fc1_b + blk * HID, f1);
    k_fc2<<<dim3(TOK / 128, 2), 256, 0, stream>>>(
        f1, wt_fc2 + (size_t)blk * CC * HID, fc2_b + blk * CC, x);
  }
}
